// CRF_9431748182048
// MI455X (gfx1250) — matrix operand order repaired and checked by reading, not compiled, not run on hardware
//
#include <hip/hip_runtime.h>
#include <hip/hip_bf16.h>
#include <math.h>

#define BATCH 256
#define NN    256
#define EE    512
#define ITERS 10

typedef _Float16 v16h __attribute__((ext_vector_type(16)));
typedef _Float16 h8   __attribute__((ext_vector_type(8)));
typedef float    v8f  __attribute__((ext_vector_type(8)));

// ---------------------------------------------------------------------------
// Kernel 1: inv_norm[b,n] = 1/||feats[b,n,:]||   (one wave per row of 512 f32)
// ---------------------------------------------------------------------------
__global__ __launch_bounds__(256)
void crf_inv_norm(const float* __restrict__ feats, float* __restrict__ inv_norm)
{
    const int lane = threadIdx.x & 31;
    const int row  = blockIdx.x * 8 + (threadIdx.x >> 5);   // 0 .. B*N-1
    const float4* p = (const float4*)(feats + (size_t)row * EE);
    float s = 0.f;
#pragma unroll
    for (int j = 0; j < 4; ++j) {
        float4 f = p[lane + 32 * j];
        s += f.x * f.x + f.y * f.y + f.z * f.z + f.w * f.w;
    }
#pragma unroll
    for (int off = 16; off > 0; off >>= 1) s += __shfl_xor(s, off, 32);
    if (lane == 0) inv_norm[row] = 1.0f / sqrtf(s);
}

// ---------------------------------------------------------------------------
// Kernel 2: wsym[n,m] = 0.5*(W[n,m] + W[m,n])
// ---------------------------------------------------------------------------
__global__ __launch_bounds__(256)
void crf_wsym(const float* __restrict__ W, float* __restrict__ wsym)
{
    const int idx = blockIdx.x * 256 + threadIdx.x;   // 0 .. N*N-1
    const int n = idx >> 8;
    const int m = idx & 255;
    wsym[idx] = 0.5f * (W[n * NN + m] + W[m * NN + n]);
}

// ---------------------------------------------------------------------------
// Kernel 3: P[b] = (F_b F_b^T) * invn invn^T * wsym   via f16 WMMA, f32 acc.
// Block = 512 threads (16 waves) computes a 128x128 tile of one batch.
// Wave grid 4x4; each wave computes a 32x32 register tile (2x2 WMMA tiles),
// so each A/B fragment feeds two WMMAs. LDS panels are double-buffered:
// one barrier per k-step, next panel's global loads overlap the WMMAs.
// ---------------------------------------------------------------------------
__global__ __launch_bounds__(512)
void crf_pairwise(const float* __restrict__ feats,
                  const float* __restrict__ inv_norm,
                  const float* __restrict__ wsym,
                  float* __restrict__ P)
{
    __shared__ _Float16 As[2][128][32];   // row panels (f16), 16 KB
    __shared__ _Float16 Bs[2][128][32];   // col panels (f16), 16 KB

    const int b       = blockIdx.z;
    const int rowBase = blockIdx.y * 128;
    const int colBase = blockIdx.x * 128;
    const int tid  = threadIdx.x;
    const int wave = tid >> 5;
    const int lane = tid & 31;
    const int wr = wave >> 2;         // wave's 32-row strip within 128 (0..3)
    const int wc = wave & 3;          // wave's 32-col strip within 128 (0..3)
    const int m  = lane & 15;         // lane's M (A) / N (B,D) index
    const int g  = lane >> 4;         // lane half-group

    const float* Fb = feats + (size_t)b * NN * EE;

    // cooperative-load coords: each thread loads 8 consecutive floats per panel
    const int lr = tid >> 2;          // 0..127 panel row
    const int lk = (tid & 3) << 3;    // 0,8,16,24 k offset

    v8f acc[2][2];
#pragma unroll
    for (int i = 0; i < 2; ++i)
#pragma unroll
        for (int j = 0; j < 2; ++j)
            acc[i][j] = (v8f){0.f, 0.f, 0.f, 0.f, 0.f, 0.f, 0.f, 0.f};

    // prefetch k0 = 0 panels into registers
    float4 pa0 = *(const float4*)(Fb + (size_t)(rowBase + lr) * EE + lk);
    float4 pa1 = *(const float4*)(Fb + (size_t)(rowBase + lr) * EE + lk + 4);
    float4 pb0 = *(const float4*)(Fb + (size_t)(colBase + lr) * EE + lk);
    float4 pb1 = *(const float4*)(Fb + (size_t)(colBase + lr) * EE + lk + 4);

    int buf = 0;
    for (int step = 0; step < EE / 32; ++step) {
        // pack f32 -> f16 and store this step's panels into LDS[buf]
        {
            h8 pk;
            pk[0] = (_Float16)pa0.x; pk[1] = (_Float16)pa0.y;
            pk[2] = (_Float16)pa0.z; pk[3] = (_Float16)pa0.w;
            pk[4] = (_Float16)pa1.x; pk[5] = (_Float16)pa1.y;
            pk[6] = (_Float16)pa1.z; pk[7] = (_Float16)pa1.w;
            *(h8*)&As[buf][lr][lk] = pk;
            pk[0] = (_Float16)pb0.x; pk[1] = (_Float16)pb0.y;
            pk[2] = (_Float16)pb0.z; pk[3] = (_Float16)pb0.w;
            pk[4] = (_Float16)pb1.x; pk[5] = (_Float16)pb1.y;
            pk[6] = (_Float16)pb1.z; pk[7] = (_Float16)pb1.w;
            *(h8*)&Bs[buf][lr][lk] = pk;
        }
        __syncthreads();

        // issue next step's global loads; latency hidden under the WMMAs
        if (step + 1 < EE / 32) {
            const int k0 = (step + 1) * 32;
            pa0 = *(const float4*)(Fb + (size_t)(rowBase + lr) * EE + k0 + lk);
            pa1 = *(const float4*)(Fb + (size_t)(rowBase + lr) * EE + k0 + lk + 4);
            pb0 = *(const float4*)(Fb + (size_t)(colBase + lr) * EE + k0 + lk);
            pb1 = *(const float4*)(Fb + (size_t)(colBase + lr) * EE + k0 + lk + 4);
        }

        // build fragments (ISA 7.12.2 layouts) and do 2x2 WMMAs
        v16h av[2], bv[2];
#pragma unroll
        for (int i = 0; i < 2; ++i) {
            const h8 alo = *(const h8*)&As[buf][wr * 32 + i * 16 + m][g * 8];
            const h8 ahi = *(const h8*)&As[buf][wr * 32 + i * 16 + m][16 + g * 8];
            const h8 blo = *(const h8*)&Bs[buf][wc * 32 + i * 16 + m][g * 8];
            const h8 bhi = *(const h8*)&Bs[buf][wc * 32 + i * 16 + m][16 + g * 8];
#pragma unroll
            for (int t = 0; t < 8; ++t) {
                av[i][t] = alo[t]; av[i][t + 8] = ahi[t];
                bv[i][t] = blo[t]; bv[i][t + 8] = bhi[t];
            }
        }
#pragma unroll
        for (int i = 0; i < 2; ++i)
#pragma unroll
            for (int j = 0; j < 2; ++j)
                acc[i][j] = __builtin_amdgcn_wmma_f32_16x16x32_f16(
                    false, av[i], false, bv[j], (short)0, acc[i][j], false, false);

        buf ^= 1;
    }

    // Epilogue: C/D layout — lane L covers N = L&15, M = 8*(L>>4) + r
#pragma unroll
    for (int j = 0; j < 2; ++j) {
        const int col  = colBase + wc * 32 + j * 16 + m;
        const float ic = inv_norm[b * NN + col];
#pragma unroll
        for (int i = 0; i < 2; ++i) {
#pragma unroll
            for (int r = 0; r < 8; ++r) {
                const int row = rowBase + wr * 32 + i * 16 + g * 8 + r;
                const float val =
                    acc[i][j][r] * inv_norm[b * NN + row] * ic * wsym[row * NN + col];
                P[((size_t)b * NN + row) * NN + col] = val;
            }
        }
    }
}

// ---------------------------------------------------------------------------
// Kernel 4: one CRF mean-field step.  e[b,n] = sum_m P[b,n,m]*(2*sigm(lg[b,m])-1)
// lg_out = logits0 + e.  One workgroup per batch; P streams from L2 (67MB<192MB).
// ---------------------------------------------------------------------------
__global__ __launch_bounds__(256)
void crf_iter(const float* __restrict__ P,
              const float* __restrict__ lg_in,
              const float* __restrict__ logits0,
              float* __restrict__ lg_out)
{
    __shared__ float v[NN];
    const int b   = blockIdx.x;
    const int tid = threadIdx.x;
    {
        const float x = lg_in[b * NN + tid];
        v[tid] = 2.0f / (1.0f + expf(-x)) - 1.0f;
    }
    __syncthreads();

    const int wave = tid >> 5;
    const int lane = tid & 31;
    for (int row = wave; row < NN; row += 8) {
        const float* pr = P + ((size_t)b * NN + row) * NN;
        float s = 0.f;
#pragma unroll
        for (int t = 0; t < NN / 32; ++t)
            s += pr[lane + 32 * t] * v[lane + 32 * t];
#pragma unroll
        for (int off = 16; off > 0; off >>= 1) s += __shfl_xor(s, off, 32);
        if (lane == 0) lg_out[b * NN + row] = logits0[b * NN + row] + s;
    }
}

// ---------------------------------------------------------------------------
extern "C" void kernel_launch(void* const* d_in, const int* in_sizes, int n_in,
                              void* d_out, int out_size, void* d_ws, size_t ws_size,
                              hipStream_t stream)
{
    const float* feats  = (const float*)d_in[0];  // [B,N,E]
    const float* logits = (const float*)d_in[1];  // [B,N,1]
    const float* W      = (const float*)d_in[2];  // [1,N,N]
    float* out = (float*)d_out;                   // [B,N,1]

    char* ws = (char*)d_ws;
    float* P        = (float*)ws;                                   // 67,108,864 B
    float* inv_norm = (float*)(ws + (size_t)BATCH * NN * NN * 4);   // 262,144 B
    float* wsym     = inv_norm + BATCH * NN;                        // 262,144 B
    float* lgA      = wsym + NN * NN;                               // 262,144 B
    float* lgB      = lgA + BATCH * NN;                             // 262,144 B

    crf_inv_norm<<<BATCH * NN / 8, 256, 0, stream>>>(feats, inv_norm);
    crf_wsym<<<NN * NN / 256, 256, 0, stream>>>(W, wsym);

    dim3 grid(NN / 128, NN / 128, BATCH);   // 2 x 2 x 256 workgroups
    crf_pairwise<<<grid, 512, 0, stream>>>(feats, inv_norm, wsym, P);

    const float* cur = logits;
    for (int t = 0; t < ITERS; ++t) {
        float* o = (t == ITERS - 1) ? out : ((t & 1) ? lgB : lgA);
        crf_iter<<<BATCH, 256, 0, stream>>>(P, cur, logits, o);
        cur = o;
    }
}
